// MultimodalGNN_70695161692530
// MI455X (gfx1250) — matrix operand order repaired and checked by reading, not compiled, not run on hardware
//
#include <hip/hip_runtime.h>
#include <hip/hip_bf16.h>

// ---------------- types for WMMA fragments ----------------
typedef __attribute__((ext_vector_type(16))) __bf16        v16bf;
typedef __attribute__((ext_vector_type(8)))  float         v8f;
typedef __attribute__((ext_vector_type(4)))  unsigned int  u32x4;

union FragAB { v16bf v; u32x4 q[2]; };

#define GN  262144   // nodes
#define GE  262144   // edges
#define GF  256      // feature dim
#define GH  256      // hidden dim
#define GR  64       // relations
#define GHC 128      // classifier hidden
#define GC  2        // classes

static __device__ __forceinline__ unsigned short f32_to_bf16(float f) {
    unsigned int u = __float_as_uint(f);
    unsigned int r = (u + 0x7FFFu + ((u >> 16) & 1u)) >> 16;  // RNE
    return (unsigned short)r;
}

// ---------------- elementwise / prep kernels ----------------
__global__ void k_zero_i32(int* p, int n) {
    int i = blockIdx.x * blockDim.x + threadIdx.x;
    if (i < n) p[i] = 0;
}

__global__ void k_degree(const int* __restrict__ dst, int* __restrict__ deg, int e) {
    int i = blockIdx.x * blockDim.x + threadIdx.x;
    if (i < e) atomicAdd(&deg[dst[i]], 1);
}

__global__ void k_dinv(const int* __restrict__ deg, float* __restrict__ dinv, int n) {
    int i = blockIdx.x * blockDim.x + threadIdx.x;
    if (i < n) dinv[i] = rsqrtf((float)(deg[i] + 1));  // +1 self-loop
}

__global__ void k_cvt_bf16(const float* __restrict__ in, unsigned short* __restrict__ out, int n) {
    int i = blockIdx.x * blockDim.x + threadIdx.x;
    if (i < n) out[i] = f32_to_bf16(in[i]);
}

// W: [K, Nout] fp32 row-major -> Wt: [Nout, K] bf16 row-major
__global__ void k_transpose_cvt(const float* __restrict__ W, unsigned short* __restrict__ Wt,
                                int K, int Nout) {
    int i = blockIdx.x * blockDim.x + threadIdx.x;
    if (i >= K * Nout) return;
    int nrow = i / K;
    int k    = i - nrow * K;
    Wt[i] = f32_to_bf16(W[(size_t)k * Nout + nrow]);
}

// ---------------- WMMA GEMM: C[M,Nout] = A[M,K](bf16) * Bt[Nout,K](bf16)^T ----------------
// Each wave computes a 16x64 output strip: 4 independent v_wmma_f32_16x16x32_bf16
// accumulator chains (pipeline-friendly), A fragment reused 4x per K-step.
// Requires Nout % 64 == 0 and M % 16 == 0.
__global__ __launch_bounds__(256)
void k_gemm_wmma(const unsigned short* __restrict__ A,
                 const unsigned short* __restrict__ Bt,
                 float* __restrict__ C,
                 const float* __restrict__ bias,   // may be null
                 int M, int K, int Nout, int relu) {
    const int wave    = blockIdx.x * (blockDim.x >> 5) + (threadIdx.x >> 5);
    const int lane    = threadIdx.x & 31;
    const int half    = lane >> 4;      // 0 or 1
    const int lm      = lane & 15;
    const int tiles_n = Nout >> 6;      // 64-wide strips
    const int tm      = wave / tiles_n; // wave-uniform
    const int tng     = wave - tm * tiles_n;
    const int n_base  = tng << 6;
    if (tm * 16 >= M) return;

    // A lane layout (16-bit A, 16x32): contiguous K {half*16 .. +16}
    const unsigned short* arow = A + (size_t)(tm * 16 + lm) * K + half * 16;
    // B lane layout (16-bit B, 32x16): contiguous K {half*16 .. +16} for N = n_base+j*16+lm
    const unsigned short* brow0 = Bt + (size_t)(n_base + lm) * K + half * 16;

    v8f c0 = {}, c1 = {}, c2 = {}, c3 = {};
    for (int k0 = 0; k0 < K; k0 += 32) {
        FragAB fa;
        const u32x4* pa = (const u32x4*)(arow + k0);
        fa.q[0] = pa[0];          // 8 bf16: K = k0+half*16 .. +8
        fa.q[1] = pa[1];          // 8 bf16: K = k0+half*16+8 .. +8

        FragAB fb0, fb1, fb2, fb3;
        const u32x4* pb0 = (const u32x4*)(brow0 + k0);
        const u32x4* pb1 = (const u32x4*)(brow0 + (size_t)16 * K + k0);
        const u32x4* pb2 = (const u32x4*)(brow0 + (size_t)32 * K + k0);
        const u32x4* pb3 = (const u32x4*)(brow0 + (size_t)48 * K + k0);
        fb0.q[0] = pb0[0]; fb0.q[1] = pb0[1];
        fb1.q[0] = pb1[0]; fb1.q[1] = pb1[1];
        fb2.q[0] = pb2[0]; fb2.q[1] = pb2[1];
        fb3.q[0] = pb3[0]; fb3.q[1] = pb3[1];

        c0 = __builtin_amdgcn_wmma_f32_16x16x32_bf16(false, fa.v, false, fb0.v, (short)0, c0, false, false);
        c1 = __builtin_amdgcn_wmma_f32_16x16x32_bf16(false, fa.v, false, fb1.v, (short)0, c1, false, false);
        c2 = __builtin_amdgcn_wmma_f32_16x16x32_bf16(false, fa.v, false, fb2.v, (short)0, c2, false, false);
        c3 = __builtin_amdgcn_wmma_f32_16x16x32_bf16(false, fa.v, false, fb3.v, (short)0, c3, false, false);
    }

    // D layout per tile: lane => N = n_base + j*16 + lm ; element e => M = tm*16 + half*8 + e
    float* crow = C + (size_t)(tm * 16 + half * 8) * Nout + n_base + lm;
    const v8f accs[4] = {c0, c1, c2, c3};
#pragma unroll
    for (int j = 0; j < 4; ++j) {
        const int   ncol = n_base + j * 16 + lm;
        const float bv   = bias ? bias[ncol] : 0.0f;
        v8f cj = accs[j];
#pragma unroll
        for (int e = 0; e < 8; ++e) {
            float v = cj[e] + bv;
            if (relu) v = fmaxf(v, 0.0f);
            crow[(size_t)e * Nout + j * 16] = v;
        }
    }
}

// ---------------- GCN aggregation ----------------
// abuf[v][c] = dinv[v]^2 * h[v][c]   (self-loop term)
__global__ void k_selfinit(const float* __restrict__ h, const float* __restrict__ dinv,
                           float* __restrict__ abuf, int total, int Hdim) {
    int i = blockIdx.x * blockDim.x + threadIdx.x;
    if (i >= total) return;
    int row = i / Hdim;
    float dv = dinv[row];
    abuf[i] = dv * dv * h[i];
}

// one block per edge, one thread per column: abuf[dst] += dinv[s]*dinv[d]*h[src]
__global__ __launch_bounds__(GH)
void k_scatter(const int* __restrict__ src, const int* __restrict__ dst,
               const float* __restrict__ dinv, const float* __restrict__ h,
               float* __restrict__ abuf) {
    int e   = blockIdx.x;
    int col = threadIdx.x;
    int s = src[e], d = dst[e];
    float nrm = dinv[s] * dinv[d];
    atomicAdd(&abuf[(size_t)d * GH + col], nrm * h[(size_t)s * GH + col]);
}

// h1 = relu(abuf + b1) + rel_emb[edge_type[dst[row]]], re-staged as bf16
__global__ void k_post1(const float* __restrict__ abuf, const float* __restrict__ b1,
                        const float* __restrict__ rel, const int* __restrict__ etype,
                        const int* __restrict__ dst, unsigned short* __restrict__ outb,
                        int total) {
    int i = blockIdx.x * blockDim.x + threadIdx.x;
    if (i >= total) return;
    int row = i >> 8;          // /GH
    int col = i & (GH - 1);
    float v = fmaxf(abuf[i] + b1[col], 0.0f);
    int r = etype[dst[row]];
    v += rel[(size_t)r * GH + col];
    outb[i] = f32_to_bf16(v);
}

// h2 = abuf + b2, staged as bf16
__global__ void k_post2(const float* __restrict__ abuf, const float* __restrict__ b2,
                        unsigned short* __restrict__ outb, int total) {
    int i = blockIdx.x * blockDim.x + threadIdx.x;
    if (i >= total) return;
    int col = i & (GH - 1);
    outb[i] = f32_to_bf16(abuf[i] + b2[col]);
}

// out[i][c] = cb2[c] + sum_k z[i][k] * cW2[k][c]
__global__ void k_final(const float* __restrict__ z, const float* __restrict__ cW2,
                        const float* __restrict__ cb2, float* __restrict__ out, int n) {
    int i = blockIdx.x * blockDim.x + threadIdx.x;
    if (i >= n) return;
    float a0 = cb2[0], a1 = cb2[1];
    const float* zr = z + (size_t)i * GHC;
#pragma unroll 4
    for (int k = 0; k < GHC; ++k) {
        float zv = zr[k];
        a0 = fmaf(zv, cW2[k * GC + 0], a0);
        a1 = fmaf(zv, cW2[k * GC + 1], a1);
    }
    out[(size_t)i * GC + 0] = a0;
    out[(size_t)i * GC + 1] = a1;
}

// ---------------- host side ----------------
extern "C" void kernel_launch(void* const* d_in, const int* in_sizes, int n_in,
                              void* d_out, int out_size, void* d_ws, size_t ws_size,
                              hipStream_t stream) {
    const float* x     = (const float*)d_in[0];
    const int*   eidx  = (const int*)  d_in[1];   // [2,E]: src then dst
    const int*   etype = (const int*)  d_in[2];   // [E]
    const float* rel   = (const float*)d_in[3];   // [R,H]
    const float* W1    = (const float*)d_in[4];
    const float* b1    = (const float*)d_in[5];
    const float* W2    = (const float*)d_in[6];
    const float* b2    = (const float*)d_in[7];
    const float* cW1   = (const float*)d_in[8];
    const float* cb1   = (const float*)d_in[9];
    const float* cW2   = (const float*)d_in[10];
    const float* cb2   = (const float*)d_in[11];
    float* out = (float*)d_out;

    const int* src = eidx;
    const int* dst = eidx + GE;

    // workspace layout
    char* ws = (char*)d_ws;
    int*            deg  = (int*)ws;                                    // N ints
    float*          dinv = (float*)(ws + ((size_t)1 << 20));            // N floats
    unsigned short* W1t  = (unsigned short*)(ws + ((size_t)2 << 20));   // [H,F] bf16
    unsigned short* W2t  = W1t + GH * GF;                               // [H,H] bf16
    unsigned short* cW1t = W2t + GH * GH;                               // [128,256] bf16
    float*          hbuf = (float*)(ws + ((size_t)4 << 20));            // [N,H] f32
    float*          abuf = hbuf + (size_t)GN * GH;                      // [N,H] f32
    unsigned short* xb   = (unsigned short*)(abuf + (size_t)GN * GH);   // [N,H] bf16
    float*          zf   = hbuf;                                        // reuse for z [N,128]

    const int T = 256;
    const int totNH = GN * GH;

    // 1) degrees and dinv
    k_zero_i32<<<(GN + T - 1) / T, T, 0, stream>>>(deg, GN);
    k_degree  <<<(GE + T - 1) / T, T, 0, stream>>>(dst, deg, GE);
    k_dinv    <<<(GN + T - 1) / T, T, 0, stream>>>(deg, dinv, GN);

    // 2) stage x and weights in bf16 (weights transposed to [Nout,K])
    k_cvt_bf16     <<<(totNH + T - 1) / T, T, 0, stream>>>(x, xb, totNH);
    k_transpose_cvt<<<(GF * GH  + T - 1) / T, T, 0, stream>>>(W1,  W1t,  GF, GH);
    k_transpose_cvt<<<(GH * GH  + T - 1) / T, T, 0, stream>>>(W2,  W2t,  GH, GH);
    k_transpose_cvt<<<(GH * GHC + T - 1) / T, T, 0, stream>>>(cW1, cW1t, GH, GHC);

    // 3) conv1: GEMM -> self-loop init -> edge scatter -> bias/relu/rel_emb fuse
    {
        int waves = (GN / 16) * (GH / 64);   // 16x64 strip per wave
        k_gemm_wmma<<<waves / 8, 256, 0, stream>>>(xb, W1t, hbuf, nullptr, GN, GF, GH, 0);
    }
    k_selfinit<<<(totNH + T - 1) / T, T, 0, stream>>>(hbuf, dinv, abuf, totNH, GH);
    k_scatter <<<GE, GH, 0, stream>>>(src, dst, dinv, hbuf, abuf);
    k_post1   <<<(totNH + T - 1) / T, T, 0, stream>>>(abuf, b1, rel, etype, dst, xb, totNH);

    // 4) conv2
    {
        int waves = (GN / 16) * (GH / 64);
        k_gemm_wmma<<<waves / 8, 256, 0, stream>>>(xb, W2t, hbuf, nullptr, GN, GH, GH, 0);
    }
    k_selfinit<<<(totNH + T - 1) / T, T, 0, stream>>>(hbuf, dinv, abuf, totNH, GH);
    k_scatter <<<GE, GH, 0, stream>>>(src, dst, dinv, hbuf, abuf);
    k_post2   <<<(totNH + T - 1) / T, T, 0, stream>>>(abuf, b2, xb, totNH);

    // 5) classifier: z = relu(h2 @ cW1 + cb1)  (fused bias+relu in GEMM)
    {
        int waves = (GN / 16) * (GHC / 64);
        k_gemm_wmma<<<waves / 8, 256, 0, stream>>>(xb, cW1t, zf, cb1, GN, GH, GHC, 1);
    }
    // 6) out = z @ cW2 + cb2
    k_final<<<(GN + T - 1) / T, T, 0, stream>>>(zf, cW2, cb2, out, GN);
}
